// TreeToSequence_7988639171346
// MI455X (gfx1250) — matrix operand order repaired and checked by reading, not compiled, not run on hardware
//
#include <hip/hip_runtime.h>
#include <stdint.h>

// ---------------------------------------------------------------------------
// Problem constants
// ---------------------------------------------------------------------------
#define NCLASS 50000
#define NOUT   (NCLASS + 1)     // 50001 rows in W_out
#define HIDDEN 1024
#define GATES  (4 * HIDDEN)     // 4096
#define EMBED  512
#define SEQ    512

#define BN_CLS 64               // classes per block (4 n-tiles)
#define NBLK   782              // ceil(50001/64)
#define NPAD   (NBLK * BN_CLS)  // 50048 padded classes

typedef __attribute__((ext_vector_type(16))) __bf16 v16bf;
typedef __attribute__((ext_vector_type(8)))  float  v8f;

// vector-of-int types matching the async-to-LDS builtin signatures
typedef int v4i __attribute__((vector_size(16)));
typedef __attribute__((address_space(1))) v4i* gptr4;
typedef __attribute__((address_space(3))) v4i* lptr4;

// ---------------------------------------------------------------------------
// CDNA5 async global->LDS copy (ASYNCcnt path), guarded with fallback
// ---------------------------------------------------------------------------
#if __has_builtin(__builtin_amdgcn_global_load_async_to_lds_b128)
#define USE_ASYNC_LDS 1
#endif

static __device__ __forceinline__ void cp_b128(const uint16_t* g, uint16_t* l) {
#ifdef USE_ASYNC_LDS
    __builtin_amdgcn_global_load_async_to_lds_b128((gptr4)g, (lptr4)l, 0, 0);
#else
    *(uint4*)l = *(const uint4*)g;
#endif
}
// Allow up to the 2 just-issued (next-stage) async ops to remain outstanding.
static __device__ __forceinline__ void wait_async_prev_stage() {
#ifdef USE_ASYNC_LDS
# if __has_builtin(__builtin_amdgcn_s_wait_asynccnt)
    __builtin_amdgcn_s_wait_asynccnt(2);
# else
    asm volatile("s_wait_asynccnt 0x2" ::: "memory");
# endif
#endif
}
static __device__ __forceinline__ void wait_async_all() {
#ifdef USE_ASYNC_LDS
# if __has_builtin(__builtin_amdgcn_s_wait_asynccnt)
    __builtin_amdgcn_s_wait_asynccnt(0);
# else
    asm volatile("s_wait_asynccnt 0x0" ::: "memory");
# endif
#endif
}

// ---------------------------------------------------------------------------
// bf16 helpers (bit-level)
// ---------------------------------------------------------------------------
static __device__ __forceinline__ uint16_t f2bf(float f) {
    uint32_t u = __float_as_uint(f);
    u += 0x7FFFu + ((u >> 16) & 1u);          // round-to-nearest-even
    return (uint16_t)(u >> 16);
}
static __device__ __forceinline__ float bf2f(uint16_t h) {
    return __uint_as_float(((uint32_t)h) << 16);
}

static __device__ __forceinline__ float sigm(float x) {
    return 1.0f / (1.0f + __expf(-x));
}

static __device__ __forceinline__ uint4 pack8(const float* p) {   // 8 f32 -> 8 bf16
    float4 a = ((const float4*)p)[0];
    float4 b = ((const float4*)p)[1];
    uint4 r;
    r.x = (uint32_t)f2bf(a.x) | ((uint32_t)f2bf(a.y) << 16);
    r.y = (uint32_t)f2bf(a.z) | ((uint32_t)f2bf(a.w) << 16);
    r.z = (uint32_t)f2bf(b.x) | ((uint32_t)f2bf(b.y) << 16);
    r.w = (uint32_t)f2bf(b.z) | ((uint32_t)f2bf(b.w) << 16);
    return r;
}

// ---------------------------------------------------------------------------
// Prep kernels (vectorized: these replay inside the timed graph)
// ---------------------------------------------------------------------------
__global__ void k_prep_wout(const float* __restrict__ W, uint16_t* __restrict__ Wb) {
    long g = (long)blockIdx.x * blockDim.x + threadIdx.x;     // groups of 8
    if (g >= (long)NPAD * HIDDEN / 8) return;
    int n  = (int)(g / (HIDDEN / 8));
    int k8 = (int)(g % (HIDDEN / 8)) * 8;
    uint4 v = {0, 0, 0, 0};
    if (n < NOUT) v = pack8(&W[(size_t)n * HIDDEN + k8]);
    ((uint4*)Wb)[g] = v;
}
__global__ void k_prep_bout(const float* __restrict__ b, float* __restrict__ bp) {
    int n = blockIdx.x * blockDim.x + threadIdx.x;
    if (n >= NPAD) return;
    bp[n] = (n < NOUT) ? b[n] : -1.0e30f;                     // pads vanish in softmax
}
__global__ void k_prep_wih(const float* __restrict__ W, uint16_t* __restrict__ Wb) {
    long g = (long)blockIdx.x * blockDim.x + threadIdx.x;     // groups of 8
    if (g >= (long)GATES * EMBED / 8) return;
    ((uint4*)Wb)[g] = pack8(&W[g * 8]);
}
__global__ void k_prep_bias(const float* __restrict__ bih, const float* __restrict__ bhh,
                            float* __restrict__ bsum) {
    int n = blockIdx.x * blockDim.x + threadIdx.x;
    if (n < GATES) bsum[n] = bih[n] + bhh[n];
}
// W_hh [4096 x 1024] row-major  ->  packed [512 k-pairs][4096 rows] of bf16x2
__global__ void k_prep_whh2(const float* __restrict__ W, uint32_t* __restrict__ P) {
    long idx = (long)blockIdx.x * blockDim.x + threadIdx.x;   // over 512*4096
    if (idx >= (long)(HIDDEN / 2) * GATES) return;
    int k2 = (int)(idx / GATES), row = (int)(idx % GATES);
    float2 v = ((const float2*)W)[(size_t)row * (HIDDEN / 2) + k2];
    P[idx] = (uint32_t)f2bf(v.x) | ((uint32_t)f2bf(v.y) << 16);
}
// X[t][k] = emb[token(t)][k] bf16; token(0)=SOS=NCLASS+1
__global__ void k_prep_x(const int* __restrict__ target, const float* __restrict__ emb,
                         uint16_t* __restrict__ X) {
    long g = (long)blockIdx.x * blockDim.x + threadIdx.x;     // groups of 8
    if (g >= (long)SEQ * EMBED / 8) return;
    int t  = (int)(g / (EMBED / 8));
    int k8 = (int)(g % (EMBED / 8)) * 8;
    int tok = (t == 0) ? (NCLASS + 1) : target[t - 1];
    ((uint4*)X)[g] = pack8(&emb[(size_t)tok * EMBED + k8]);
}

// ---------------------------------------------------------------------------
// WMMA fragment loaders (CDNA5 wave32 layouts, 16x16x32 bf16).
// Each fragment is exactly two contiguous 16B chunks per lane -> force
// explicit 128-bit LDS loads (ds_load_b128) via union bitcast.
//   A 16x32: m = lane%16, grp = lane/16; i=0..7 -> k=grp*8+i, i=8..15 -> k=16+grp*8+(i-8)
//   B 32x16: n = lane%16, grp = lane/16; i=0..7 -> k=grp*8+i, i=8..15 -> k=16+grp*8+(i-8)
// ---------------------------------------------------------------------------
union V16U { v16bf v; uint4 q[2]; };

static __device__ __forceinline__ v16bf load_afrag(const uint16_t* As, int mt, int lane) {
    int grp = lane >> 4, mr = lane & 15;
    const uint16_t* base = &As[(mt * 16 + mr) * 32];
    V16U u;
    u.q[0] = *(const uint4*)(base + grp * 8);         // k = grp*8 .. grp*8+7
    u.q[1] = *(const uint4*)(base + 16 + grp * 8);    // k = 16+grp*8 .. 23+grp*8
    return u.v;
}
static __device__ __forceinline__ v16bf load_bfrag(const uint16_t* Bs, int nt, int lane) {
    int grp = lane >> 4, nr = lane & 15;
    const uint16_t* base = &Bs[(nt * 16 + nr) * 32];
    V16U u;
    u.q[0] = *(const uint4*)(base + grp * 8);         // k = grp*8 .. grp*8+7
    u.q[1] = *(const uint4*)(base + 16 + grp * 8);    // k = 16+grp*8 .. 23+grp*8
    return u.v;
}

// ---------------------------------------------------------------------------
// Shared GEMM core: BM=64 x BN=64, BK=32, 256 threads (8 waves), bf16 WMMA,
// double-buffered LDS fed by async global->LDS copies. Each wave computes
// TWO 16x16 output tiles sharing one A fragment (2 WMMA per K-chunk).
//   A: each thread copies one 16B chunk (64*32*2B = 4096B = 256*16B)
//   B: each thread copies one 16B chunk (64*32*2B = 4096B = 256*16B)
// ---------------------------------------------------------------------------
#define GEMM_STAGE(buf, k0)                                                      \
    do {                                                                         \
        int r_ = tid >> 2, kc_ = (tid & 3) * 8;                                  \
        cp_b128(&A[(size_t)(m0 + r_) * K + (k0) + kc_], &As[buf][r_ * 32 + kc_]);\
        cp_b128(&B[(size_t)(n0 + r_) * K + (k0) + kc_], &Bs[buf][r_ * 32 + kc_]);\
    } while (0)

// ---------------------------------------------------------------------------
// GEMM 1: Gih[512 x 4096] = X[512x512] @ W_ih^T + (b_ih + b_hh)
// ---------------------------------------------------------------------------
__global__ __launch_bounds__(256) void k_gemm_gih(const uint16_t* __restrict__ A,
                                                  const uint16_t* __restrict__ B,
                                                  const float* __restrict__ bias,
                                                  float* __restrict__ out) {
    const int K = EMBED;
    __shared__ __align__(16) uint16_t As[2][64 * 32];
    __shared__ __align__(16) uint16_t Bs[2][64 * 32];
    const int tid = threadIdx.x;
    const int m0 = blockIdx.x * 64;
    const int n0 = blockIdx.y * 64;
    const int wave = tid >> 5, lane = tid & 31;
    const int mt = wave & 3;                 // 4 m-tiles
    const int ntA = (wave >> 2) * 2;         // 2 n-tiles per wave: ntA, ntA+1
    v8f acc0 = {}, acc1 = {};
    GEMM_STAGE(0, 0);
    const int NK = K / 32;
    for (int kc = 0; kc < NK; ++kc) {
        if (kc + 1 < NK) GEMM_STAGE((kc + 1) & 1, (kc + 1) * 32);
        wait_async_prev_stage();
        __syncthreads();
        v16bf af = load_afrag(As[kc & 1], mt, lane);
        v16bf b0 = load_bfrag(Bs[kc & 1], ntA, lane);
        v16bf b1 = load_bfrag(Bs[kc & 1], ntA + 1, lane);
        acc0 = __builtin_amdgcn_wmma_f32_16x16x32_bf16(false, af, false, b0,
                                                       (short)0, acc0, false, false);
        acc1 = __builtin_amdgcn_wmma_f32_16x16x32_bf16(false, af, false, b1,
                                                       (short)0, acc1, false, false);
        __syncthreads();
    }
    wait_async_all();
    const int grp = lane >> 4, lr = lane & 15;
    const int na = n0 + ntA * 16 + lr;
    const int nb = n0 + (ntA + 1) * 16 + lr;
#pragma unroll
    for (int v = 0; v < 8; ++v) {
        int m = m0 + mt * 16 + v + 8 * grp;
        out[(size_t)m * GATES + na] = acc0[v] + bias[na];
        out[(size_t)m * GATES + nb] = acc1[v] + bias[nb];
    }
}

// ---------------------------------------------------------------------------
// Sequential LSTM scan: 1 workgroup, 1024 threads, 512 steps.
// ---------------------------------------------------------------------------
__global__ __launch_bounds__(1024) void k_scan(const float* __restrict__ Gih,
                                               const uint32_t* __restrict__ Whh2,
                                               const float* __restrict__ h0,
                                               const float* __restrict__ c0,
                                               uint16_t* __restrict__ H2) {
    __shared__ float hb[2][HIDDEN];
    const int j = threadIdx.x;
    hb[0][j] = h0[j];
    float c = c0[j];
    __syncthreads();
    for (int t = 0; t < SEQ; ++t) {
        const float* hp = hb[t & 1];
        float a0 = Gih[(size_t)t * GATES + j];
        float a1 = Gih[(size_t)t * GATES + HIDDEN + j];
        float a2 = Gih[(size_t)t * GATES + 2 * HIDDEN + j];
        float a3 = Gih[(size_t)t * GATES + 3 * HIDDEN + j];
#pragma unroll 4
        for (int k2 = 0; k2 < HIDDEN / 2; ++k2) {
            float hlo = hp[2 * k2], hhi = hp[2 * k2 + 1];
            size_t base = (size_t)k2 * GATES + j;
            uint32_t wa = Whh2[base];
            uint32_t wb = Whh2[base + HIDDEN];
            uint32_t wc = Whh2[base + 2 * HIDDEN];
            uint32_t wd = Whh2[base + 3 * HIDDEN];
            a0 += __uint_as_float(wa << 16) * hlo + __uint_as_float(wa & 0xFFFF0000u) * hhi;
            a1 += __uint_as_float(wb << 16) * hlo + __uint_as_float(wb & 0xFFFF0000u) * hhi;
            a2 += __uint_as_float(wc << 16) * hlo + __uint_as_float(wc & 0xFFFF0000u) * hhi;
            a3 += __uint_as_float(wd << 16) * hlo + __uint_as_float(wd & 0xFFFF0000u) * hhi;
        }
        float ig = sigm(a0), fg = sigm(a1), gg = tanhf(a2), og = sigm(a3);
        c = fg * c + ig * gg;
        float h2 = og * tanhf(c);
        hb[(t + 1) & 1][j] = h2;
        H2[(size_t)t * HIDDEN + j] = f2bf(h2);
        __syncthreads();
    }
}

// ---------------------------------------------------------------------------
// GEMM 2 + fused online-softmax partials: logits = H2 @ W_out^T + b_out,
// reduced per 64x64 block into per-row (max, sum-exp) partials. K=1024.
// ---------------------------------------------------------------------------
__global__ __launch_bounds__(256) void k_gemm_logits(const uint16_t* __restrict__ A,
                                                     const uint16_t* __restrict__ B,
                                                     const float* __restrict__ bias,
                                                     float* __restrict__ pmax,
                                                     float* __restrict__ psum) {
    const int K = HIDDEN;
    __shared__ __align__(16) uint16_t As[2][64 * 32];
    __shared__ __align__(16) uint16_t Bs[2][64 * 32];
    __shared__ float Cs[64 * BN_CLS];
    const int tid = threadIdx.x;
    const int m0 = blockIdx.x * 64;
    const int n0 = blockIdx.y * BN_CLS;
    const int wave = tid >> 5, lane = tid & 31;
    const int mt = wave & 3;
    const int ntA = (wave >> 2) * 2;
    v8f acc0 = {}, acc1 = {};
    GEMM_STAGE(0, 0);
    const int NK = K / 32;
    for (int kc = 0; kc < NK; ++kc) {
        if (kc + 1 < NK) GEMM_STAGE((kc + 1) & 1, (kc + 1) * 32);
        wait_async_prev_stage();
        __syncthreads();
        v16bf af = load_afrag(As[kc & 1], mt, lane);
        v16bf b0 = load_bfrag(Bs[kc & 1], ntA, lane);
        v16bf b1 = load_bfrag(Bs[kc & 1], ntA + 1, lane);
        acc0 = __builtin_amdgcn_wmma_f32_16x16x32_bf16(false, af, false, b0,
                                                       (short)0, acc0, false, false);
        acc1 = __builtin_amdgcn_wmma_f32_16x16x32_bf16(false, af, false, b1,
                                                       (short)0, acc1, false, false);
        __syncthreads();
    }
    wait_async_all();
    const int grp = lane >> 4, lr = lane & 15;
    const int ca = ntA * 16 + lr;
    const int cb = (ntA + 1) * 16 + lr;
#pragma unroll
    for (int v = 0; v < 8; ++v) {
        int m = mt * 16 + v + 8 * grp;
        Cs[m * BN_CLS + ca] = acc0[v] + bias[n0 + ca];
        Cs[m * BN_CLS + cb] = acc1[v] + bias[n0 + cb];
    }
    __syncthreads();
    if (tid < 64) {
        float mx = -3.0e38f;
#pragma unroll 4
        for (int n = 0; n < BN_CLS; ++n) mx = fmaxf(mx, Cs[tid * BN_CLS + n]);
        float s = 0.0f;
#pragma unroll 4
        for (int n = 0; n < BN_CLS; ++n) s += __expf(Cs[tid * BN_CLS + n] - mx);
        size_t o = (size_t)(m0 + tid) * NBLK + blockIdx.y;
        pmax[o] = mx;
        psum[o] = s;
    }
}

// ---------------------------------------------------------------------------
// Target logit per step: tl[t] = h2[t] . W_out[target[t]] + b_out[target[t]]
// ---------------------------------------------------------------------------
__global__ __launch_bounds__(256) void k_tlogit(const uint16_t* __restrict__ H2,
                                                const float* __restrict__ Wout,
                                                const float* __restrict__ bout,
                                                const int* __restrict__ target,
                                                float* __restrict__ tl) {
    __shared__ float red[256];
    const int t = blockIdx.x;
    const int tg = target[t];
    float s = 0.0f;
    for (int k = threadIdx.x; k < HIDDEN; k += 256)
        s += bf2f(H2[(size_t)t * HIDDEN + k]) * Wout[(size_t)tg * HIDDEN + k];
    red[threadIdx.x] = s;
    __syncthreads();
    for (int o = 128; o > 0; o >>= 1) {
        if (threadIdx.x < o) red[threadIdx.x] += red[threadIdx.x + o];
        __syncthreads();
    }
    if (threadIdx.x == 0) tl[t] = red[0] + bout[tg];
}

// ---------------------------------------------------------------------------
// Final reduce: per-step log-sum-exp across NBLK partials, ce sum -> scalar
// ---------------------------------------------------------------------------
__global__ __launch_bounds__(256) void k_final(const float* __restrict__ pmax,
                                               const float* __restrict__ psum,
                                               const float* __restrict__ tl,
                                               float* __restrict__ out) {
    __shared__ float red[256];
    float acc = 0.0f;
    for (int t = threadIdx.x; t < SEQ; t += 256) {
        const float* pm = pmax + (size_t)t * NBLK;
        const float* ps = psum + (size_t)t * NBLK;
        float m = -3.0e38f;
        for (int b = 0; b < NBLK; ++b) m = fmaxf(m, pm[b]);
        float s = 0.0f;
        for (int b = 0; b < NBLK; ++b) s += ps[b] * __expf(pm[b] - m);
        acc += (m + __logf(s)) - tl[t];
    }
    red[threadIdx.x] = acc;
    __syncthreads();
    for (int o = 128; o > 0; o >>= 1) {
        if (threadIdx.x < o) red[threadIdx.x] += red[threadIdx.x + o];
        __syncthreads();
    }
    if (threadIdx.x == 0) out[0] = red[0];
}

// ---------------------------------------------------------------------------
// Host-side launch
// ---------------------------------------------------------------------------
extern "C" void kernel_launch(void* const* d_in, const int* in_sizes, int n_in,
                              void* d_out, int out_size, void* d_ws, size_t ws_size,
                              hipStream_t stream) {
    const int*   target = (const int*)  d_in[0];
    const float* h0     = (const float*)d_in[1];
    const float* c0     = (const float*)d_in[2];
    const float* emb    = (const float*)d_in[3];
    const float* W_ih   = (const float*)d_in[4];
    const float* W_hh   = (const float*)d_in[5];
    const float* b_ih   = (const float*)d_in[6];
    const float* b_hh   = (const float*)d_in[7];
    const float* W_out  = (const float*)d_in[8];
    const float* b_out  = (const float*)d_in[9];
    float* out = (float*)d_out;

    // workspace layout (bytes)
    char* w = (char*)d_ws;
    const size_t SZ_WOUT = (size_t)NPAD * HIDDEN * 2;     // 102.5 MB bf16 (L2 resident)
    const size_t SZ_BOUT = (size_t)NPAD * 4;
    const size_t SZ_WIH  = (size_t)GATES * EMBED * 2;
    const size_t SZ_BIAS = (size_t)GATES * 4;
    const size_t SZ_X    = (size_t)SEQ * EMBED * 2;
    const size_t SZ_GIH  = (size_t)SEQ * GATES * 4;
    const size_t SZ_WHH2 = (size_t)(HIDDEN / 2) * GATES * 4;
    const size_t SZ_H2   = (size_t)SEQ * HIDDEN * 2;
    const size_t SZ_PART = (size_t)SEQ * NBLK * 4;

    size_t off = 0;
    uint16_t* Wout_bf = (uint16_t*)(w + off); off += SZ_WOUT;
    float*    bout_p  = (float*)   (w + off); off += SZ_BOUT;
    uint16_t* Wih_bf  = (uint16_t*)(w + off); off += SZ_WIH;
    float*    bsum    = (float*)   (w + off); off += SZ_BIAS;
    uint16_t* Xb      = (uint16_t*)(w + off); off += SZ_X;
    float*    Gih     = (float*)   (w + off); off += SZ_GIH;
    uint32_t* Whh2    = (uint32_t*)(w + off); off += SZ_WHH2;
    uint16_t* H2      = (uint16_t*)(w + off); off += SZ_H2;
    float*    pmax    = (float*)   (w + off); off += SZ_PART;
    float*    psum    = (float*)   (w + off); off += SZ_PART;
    float*    tl      = (float*)   (w + off); off += SEQ * 4;
    (void)ws_size; (void)in_sizes; (void)n_in; (void)out_size;

    // --- prep / conversion ---
    {
        long n = (long)NPAD * HIDDEN / 8;
        k_prep_wout<<<(unsigned)((n + 255) / 256), 256, 0, stream>>>(W_out, Wout_bf);
    }
    k_prep_bout<<<(NPAD + 255) / 256, 256, 0, stream>>>(b_out, bout_p);
    {
        long n = (long)GATES * EMBED / 8;
        k_prep_wih<<<(unsigned)((n + 255) / 256), 256, 0, stream>>>(W_ih, Wih_bf);
    }
    k_prep_bias<<<(GATES + 255) / 256, 256, 0, stream>>>(b_ih, b_hh, bsum);
    {
        long n = (long)(HIDDEN / 2) * GATES;
        k_prep_whh2<<<(unsigned)((n + 255) / 256), 256, 0, stream>>>(W_hh, Whh2);
    }
    {
        long n = (long)SEQ * EMBED / 8;
        k_prep_x<<<(unsigned)((n + 255) / 256), 256, 0, stream>>>(target, emb, Xb);
    }

    // --- batched input-projection GEMM (WMMA) ---
    k_gemm_gih<<<dim3(SEQ / 64, GATES / 64), 256, 0, stream>>>(Xb, Wih_bf, bsum, Gih);

    // --- sequential LSTM cell scan (single WGP, 32 waves) ---
    k_scan<<<1, 1024, 0, stream>>>(Gih, Whh2, h0, c0, H2);

    // --- output projection GEMM (WMMA) fused with online-softmax partials ---
    k_gemm_logits<<<dim3(SEQ / 64, NBLK), 256, 0, stream>>>(H2, Wout_bf, bout_p, pmax, psum);

    // --- target logits + final log-sum-exp / cross-entropy reduction ---
    k_tlogit<<<SEQ, 256, 0, stream>>>(H2, W_out, b_out, target, tl);
    k_final<<<1, 256, 0, stream>>>(pmax, psum, tl, out);
}
